// MRIEncoder_37039797961209
// MI455X (gfx1250) — compile-verified, matrix operand order checked by reading, not run on hardware
//
#include <hip/hip_runtime.h>
#include <math.h>

// ---------------------------------------------------------------------------
// MRI GNN encoder (GATv2 x2 + MLP + L2-normalize) for gfx1250 (MI455X).
// Dense linear layers: v_wmma_f32_16x16x32_f16, 32x32 out per wave (4 wmma /
// K-step), double-buffered LDS staging via GLOBAL_LOAD_ASYNC_TO_LDS_B128
// (ASYNCcnt) when the builtin exists, vectorized sync staging otherwise.
// Edge attention: wave-per-(edge,head) VALU kernels; per-dst softmax via
// monotone-key atomicMax + atomicAdd (features are L2-resident: 82MB < 192MB).
// ---------------------------------------------------------------------------

typedef __attribute__((ext_vector_type(16))) _Float16 v16h;
typedef __attribute__((ext_vector_type(8)))  _Float16 v8h;
typedef __attribute__((ext_vector_type(8)))  float    v8f;
typedef __attribute__((ext_vector_type(4)))  int      v4i;

#define DEV __device__ __forceinline__

static constexpr int kH = 4;     // heads

#if defined(__HIP_DEVICE_COMPILE__) && \
    __has_builtin(__builtin_amdgcn_global_load_async_to_lds_b128) && \
    __has_builtin(__builtin_amdgcn_s_wait_asynccnt)
#define USE_ASYNC_LDS 1
#else
#define USE_ASYNC_LDS 0
#endif

// 16-byte global->LDS copy (async when available)
DEV void cp16(const _Float16* g, _Float16* l) {
#if USE_ASYNC_LDS
  __builtin_amdgcn_global_load_async_to_lds_b128((v4i*)g, (v4i*)l, 0, 0);
#else
  *(v8h*)l = *(const v8h*)g;
#endif
}
DEV void cp_wait_all() {
#if USE_ASYNC_LDS
  __builtin_amdgcn_s_wait_asynccnt(0);
#endif
}

// load a 16-wide f16 WMMA fragment from an LDS row: elements {0..7} at p[0..7],
// elements {8..15} at p[16..23] (CDNA5 16-bit operand K-layout)
DEV v16h ldfrag(const _Float16* p) {
  v16h f;
#pragma unroll
  for (int j = 0; j < 8; ++j) { f[j] = p[j]; f[j + 8] = p[j + 16]; }
  return f;
}

// ---- order-preserving float <-> uint key for atomic max ----
DEV unsigned fkey(float f) {
  unsigned b = __float_as_uint(f);
  return b ^ ((unsigned)(((int)b) >> 31) | 0x80000000u);
}
DEV float unkey(unsigned k) {
  unsigned b = (k & 0x80000000u) ? (k ^ 0x80000000u) : ~k;
  return __uint_as_float(b);
}

DEV void edge_sd(int e, int E_, const int* src, const int* dst,
                 const float* ea, const float* loop_ea,
                 int& s, int& d, float& a) {
  if (e < E_) { s = src[e]; d = dst[e]; a = ea[e]; }
  else        { s = e - E_; d = e - E_; a = loop_ea[e - E_]; }
}

// ---------------------------------------------------------------------------
__global__ void k_zero(unsigned* p, long n) {
  long i = (long)blockIdx.x * blockDim.x + threadIdx.x;
  if (i < n) p[i] = 0u;
}

// degree count + edge-attr sum per destination (self-loop fill='mean')
__global__ void k_deg(const int* dst, const float* ea, float* cnt, float* easum, int E_) {
  int e = blockIdx.x * blockDim.x + threadIdx.x;
  if (e < E_) {
    int d = dst[e];
    atomicAdd(&cnt[d], 1.0f);
    atomicAdd(&easum[d], ea[e]);
  }
}
__global__ void k_loopdiv(const float* easum, const float* cnt, float* lpe, int n) {
  int i = blockIdx.x * blockDim.x + threadIdx.x;
  if (i < n) lpe[i] = easum[i] / fmaxf(cnt[i], 1.0f);
}

// layer-1 linear: out[i,d] = b[d] + sum_{j<3} x[i,j]*W[j,d]   (K=3: VALU)
__global__ void k_lin3(const float* x, const float* W, const float* b,
                       float* out, int n, int D) {
  long idx = (long)blockIdx.x * blockDim.x + threadIdx.x;
  if (idx < (long)n * D) {
    int i = (int)(idx / D), d = (int)(idx % D);
    float a = b[d];
#pragma unroll
    for (int j = 0; j < 3; ++j) a += x[i * 3 + j] * W[j * D + d];
    out[idx] = a;
  }
}

// per-(edge,head) attention logit; wave32-parallel over channels
__global__ void __launch_bounds__(256)
k_edge_logits(const float* xl, const float* xr, const int* ei, const float* ea,
              const float* lpe, const float* We, const float* att,
              float* logit, unsigned* lmk, int E_, int Et, int C) {
  int wid  = blockIdx.x * 8 + (threadIdx.x >> 5);
  int lane = threadIdx.x & 31;
  int e = wid >> 2, h = wid & 3;
  if (e >= Et) return;
  int s, d; float av;
  edge_sd(e, E_, ei, ei + E_, ea, lpe, s, d, av);
  int D = kH * C;
  const float* pl = xl + (long)s * D + h * C;
  const float* pr = xr + (long)d * D + h * C;
  const float* pw = We + h * C;
  const float* pa = att + h * C;
  float acc = 0.f;
  for (int c = lane; c < C; c += 32) {
    float v = pl[c] + pr[c] + av * pw[c];
    v = v > 0.f ? v : 0.2f * v;                 // leaky_relu(0.2)
    acc += v * pa[c];
  }
#pragma unroll
  for (int off = 16; off > 0; off >>= 1) acc += __shfl_xor(acc, off, 32);
  if (lane == 0) {
    logit[(long)e * kH + h] = acc;
    atomicMax(&lmk[d * kH + h], fkey(acc));
  }
}

// p = exp(logit - max); accumulate denominator per (dst, head)
__global__ void k_edge_expsum(const int* ei, float* logit, const unsigned* lmk,
                              float* den, int E_, int Et) {
  long idx = (long)blockIdx.x * blockDim.x + threadIdx.x;
  if (idx >= (long)Et * kH) return;
  int e = (int)(idx >> 2), h = (int)(idx & 3);
  int d = (e < E_) ? ei[E_ + e] : (e - E_);
  float lm = unkey(lmk[d * kH + h]);
  float p = __expf(logit[idx] - lm);
  logit[idx] = p;
  atomicAdd(&den[d * kH + h], p);
}

// out[dst] += alpha * xl[src]   (wave per (edge,head))
__global__ void __launch_bounds__(256)
k_edge_scatter(const float* xl, const int* ei, const float* ea, const float* lpe,
               const float* logit, const float* den, float* out,
               int E_, int Et, int C) {
  int wid  = blockIdx.x * 8 + (threadIdx.x >> 5);
  int lane = threadIdx.x & 31;
  int e = wid >> 2, h = wid & 3;
  if (e >= Et) return;
  int s, d; float av;
  edge_sd(e, E_, ei, ei + E_, ea, lpe, s, d, av);
  int D = kH * C;
  float alpha = logit[(long)e * kH + h] / den[d * kH + h];
  const float* pl = xl + (long)s * D + h * C;
  float* po = out + (long)d * D + h * C;
  for (int c = lane; c < C; c += 32) atomicAdd(&po[c], pl[c] * alpha);
}

__global__ void k_bias_relu(float* a, const float* b, long n, int D) {
  long idx = (long)blockIdx.x * blockDim.x + threadIdx.x;
  if (idx < n) {
    float v = a[idx] + b[idx % D];
    a[idx] = fmaxf(v, 0.f);
  }
}

__global__ void k_f2h(const float* in, _Float16* out, long n) {
  long i = (long)blockIdx.x * blockDim.x + threadIdx.x;
  if (i < n) out[i] = (_Float16)in[i];
}

// convert + transpose weights: Wt[n][k] (f16) = W[k][n] (f32)
__global__ void k_f2h_t(const float* W, _Float16* Wt, int K, int Nn) {
  long idx = (long)blockIdx.x * blockDim.x + threadIdx.x;
  if (idx < (long)K * Nn) {
    int n = (int)(idx / K), k = (int)(idx % K);
    Wt[idx] = (_Float16)W[(long)k * Nn + n];
  }
}

// ---------------------------------------------------------------------------
// WMMA GEMM: C[M,N] (f32) = A[M,K](f16,row) * Bt[N,K](f16, N-major) + bias.
// Block = 256 threads = 8 waves (2 M x 4 N); block tile 64x128; each wave owns
// a 32x32 tile -> 4 v_wmma_f32_16x16x32_f16 per K-step. Double-buffered LDS,
// async global->LDS copies (ASYNCcnt), one barrier per K-step.
// Fragment mapping per CDNA5 ISA (16-bit operand 16x32 layout): lanes 0-15
// hold K in {0..7,16..23}, lanes 16-31 hold K in {8..15,24..31}.
// ---------------------------------------------------------------------------
__global__ void __launch_bounds__(256)
k_wmma_gemm(const _Float16* A, const _Float16* Bt, const float* bias, float* C,
            int M, int Nn, int K, int relu) {
  __shared__ _Float16 Ah[2][64][40];    // 64 rows x 32 k (pitch 40 = 80B)
  __shared__ _Float16 Bh[2][128][40];   // 128 n-rows x 32 k
  int t = threadIdx.x;
  int lane = t & 31, wave = t >> 5;
  int wm = wave >> 2, wn = wave & 3;
  int m0 = blockIdx.y * 64, n0 = blockIdx.x * 128;

  int mloc = wm * 32 + (lane & 15);
  int nloc = wn * 32 + (lane & 15);
  int kb = (lane >> 4) << 3;

  v8f c00 = {}, c01 = {}, c10 = {}, c11 = {};

  // stage one K-slab (32 wide) into buffer `buf`
  auto stage = [&](int buf, int k0) {
    {
      int ch = t;                       // A: 64x32 f16 = 256 x 16B chunks
      int row = ch >> 2, kc = (ch & 3) << 3;
      long gr = m0 + row;
      if (gr < M) cp16(A + gr * (long)K + k0 + kc, &Ah[buf][row][kc]);
    }
#pragma unroll
    for (int i = 0; i < 2; ++i) {       // B: 128x32 f16 = 512 x 16B chunks
      int ch = t + i * 256;
      int row = ch >> 2, kc = (ch & 3) << 3;
      cp16(Bt + (long)(n0 + row) * K + k0 + kc, &Bh[buf][row][kc]);
    }
  };

  int buf = 0;
  stage(0, 0);
  for (int k0 = 0; k0 < K; k0 += 32) {
    cp_wait_all();
    __syncthreads();                    // buf slab ready for all waves
    if (k0 + 32 < K) stage(buf ^ 1, k0 + 32);

    v16h a0 = ldfrag(&Ah[buf][mloc][kb]);
    v16h a1 = ldfrag(&Ah[buf][mloc + 16][kb]);
    v16h b0 = ldfrag(&Bh[buf][nloc][kb]);
    v16h b1 = ldfrag(&Bh[buf][nloc + 16][kb]);

    c00 = __builtin_amdgcn_wmma_f32_16x16x32_f16(false, a0, false, b0, (short)0, c00, false, false);
    c01 = __builtin_amdgcn_wmma_f32_16x16x32_f16(false, a0, false, b1, (short)0, c01, false, false);
    c10 = __builtin_amdgcn_wmma_f32_16x16x32_f16(false, a1, false, b0, (short)0, c10, false, false);
    c11 = __builtin_amdgcn_wmma_f32_16x16x32_f16(false, a1, false, b1, (short)0, c11, false, false);
    buf ^= 1;
  }

  // epilogue: C/D layout — elem r: M = r + 8*(lane>=16), N = lane&15
  int colg  = n0 + wn * 32 + (lane & 15);
  int rbase = m0 + wm * 32 + ((lane >> 4) << 3);
  auto store_tile = [&](const v8f& acc, int rofs, int cofs) {
    int col = colg + cofs;
    float bv = bias[col];
#pragma unroll
    for (int r = 0; r < 8; ++r) {
      int rowg = rbase + rofs + r;
      if (rowg < M) {
        float v = acc[r] + bv;
        if (relu) v = fmaxf(v, 0.f);
        C[(long)rowg * Nn + col] = v;
      }
    }
  };
  store_tile(c00, 0, 0);
  store_tile(c01, 0, 16);
  store_tile(c10, 16, 0);
  store_tile(c11, 16, 16);
}

// ---------------------------------------------------------------------------
__global__ void k_pool_cnt(const int* batch, float* gcnt, int n) {
  int i = blockIdx.x * blockDim.x + threadIdx.x;
  if (i < n) atomicAdd(&gcnt[batch[i]], 1.0f);
}
__global__ void k_pool_sum(const float* h, const int* batch, float* gsum, int n, int D) {
  long idx = (long)blockIdx.x * blockDim.x + threadIdx.x;
  if (idx < (long)n * D) {
    int i = (int)(idx / D), c = (int)(idx % D);
    atomicAdd(&gsum[batch[i] * D + c], h[idx]);
  }
}
// mean-pool, write padded 16-row f16 matrix for WMMA projection
__global__ void k_pool_divpad(const float* gsum, const float* gcnt,
                              _Float16* g16, int ng, int D) {
  int idx = blockIdx.x * blockDim.x + threadIdx.x;
  if (idx < 16 * D) {
    int r = idx / D, c = idx % D;
    float v = (r < ng) ? gsum[r * D + c] / fmaxf(gcnt[r], 1.0f) : 0.f;
    g16[idx] = (_Float16)v;
  }
}

// L2 normalize rows (wave per row)
__global__ void k_normalize(const float* emb, float* out, int ng, int D) {
  int wave = threadIdx.x >> 5, lane = threadIdx.x & 31;
  int r = blockIdx.x * 8 + wave;
  if (r >= ng) return;
  float ss = 0.f;
  for (int c = lane; c < D; c += 32) { float v = emb[r * D + c]; ss += v * v; }
#pragma unroll
  for (int off = 16; off > 0; off >>= 1) ss += __shfl_xor(ss, off, 32);
  float inv = 1.f / fmaxf(sqrtf(ss), 1e-12f);
  for (int c = lane; c < D; c += 32) out[r * D + c] = emb[r * D + c] * inv;
}

// ---------------------------------------------------------------------------
extern "C" void kernel_launch(void* const* d_in, const int* in_sizes, int n_in,
                              void* d_out, int out_size, void* d_ws, size_t ws_size,
                              hipStream_t stream) {
  (void)n_in; (void)out_size; (void)ws_size;
  const float* x    = (const float*)d_in[0];
  const int*   ei   = (const int*)d_in[1];      // [2,E]: src row, dst row
  const float* ea   = (const float*)d_in[2];
  const int*   bat  = (const int*)d_in[3];
  const float *Wl1 = (const float*)d_in[4],  *bl1 = (const float*)d_in[5];
  const float *Wr1 = (const float*)d_in[6],  *br1 = (const float*)d_in[7];
  const float *We1 = (const float*)d_in[8],  *at1 = (const float*)d_in[9];
  const float *b1  = (const float*)d_in[10];
  const float *Wl2 = (const float*)d_in[11], *bl2 = (const float*)d_in[12];
  const float *Wr2 = (const float*)d_in[13], *br2 = (const float*)d_in[14];
  const float *We2 = (const float*)d_in[15], *at2 = (const float*)d_in[16];
  const float *b2  = (const float*)d_in[17];
  const float *Wp1 = (const float*)d_in[18], *bp1 = (const float*)d_in[19];
  const float *Wp2 = (const float*)d_in[20], *bp2 = (const float*)d_in[21];

  const int Nn = in_sizes[0] / 3;     // 10000 nodes
  const int E_ = in_sizes[1] / 2;     // 160000 edges
  const int Et = E_ + Nn;             // + self loops
  const int D1 = 512, D2 = 1024, C1 = 128, C2 = 256, NG = 8, EMB = 1024;

  // ---- workspace bump allocator ----
  char* w = (char*)d_ws;
  size_t off = 0;
  auto alloc = [&](size_t bytes) -> void* {
    void* p = w + off; off = (off + bytes + 255) & ~(size_t)255; return p;
  };
  float*    cnt   = (float*)alloc((size_t)Nn * 4);
  float*    easum = (float*)alloc((size_t)Nn * 4);
  float*    lpe   = (float*)alloc((size_t)Nn * 4);
  float*    xl1   = (float*)alloc((size_t)Nn * D1 * 4);
  float*    xr1   = (float*)alloc((size_t)Nn * D1 * 4);
  float*    logit = (float*)alloc((size_t)Et * kH * 4);
  unsigned* lmk   = (unsigned*)alloc((size_t)Nn * kH * 4);
  float*    den   = (float*)alloc((size_t)Nn * kH * 4);
  float*    acc1  = (float*)alloc((size_t)Nn * D1 * 4);     // -> h1 in place
  _Float16* h1h   = (_Float16*)alloc((size_t)Nn * D1 * 2);
  _Float16* Wl2h  = (_Float16*)alloc((size_t)D1 * D2 * 2);  // transposed [N][K]
  _Float16* Wr2h  = (_Float16*)alloc((size_t)D1 * D2 * 2);
  _Float16* Wp1h  = (_Float16*)alloc((size_t)1024 * 1024 * 2);
  _Float16* Wp2h  = (_Float16*)alloc((size_t)1024 * 1024 * 2);
  float*    xl2   = (float*)alloc((size_t)Nn * D2 * 4);
  float*    xr2   = (float*)alloc((size_t)Nn * D2 * 4);
  float*    acc2  = (float*)alloc((size_t)Nn * D2 * 4);     // -> h2 in place
  float*    gsum  = (float*)alloc((size_t)NG * D2 * 4);
  float*    gcnt  = (float*)alloc((size_t)NG * 4);
  _Float16* g16   = (_Float16*)alloc((size_t)16 * 1024 * 2);
  float*    t1    = (float*)alloc((size_t)16 * 1024 * 4);
  _Float16* t1h   = (_Float16*)alloc((size_t)16 * 1024 * 2);
  float*    embb  = (float*)alloc((size_t)16 * 1024 * 4);

  auto cdiv = [](long a, long b) { return (int)((a + b - 1) / b); };
  auto zero = [&](void* p, long n_u32) {
    k_zero<<<cdiv(n_u32, 256), 256, 0, stream>>>((unsigned*)p, n_u32);
  };

  const long nwav = (long)Et * kH;          // waves for edge kernels
  const int  ewb  = cdiv(nwav, 8);          // 8 waves / block

  // ---- self-loop mean edge attr ----
  zero(cnt, Nn); zero(easum, Nn);
  k_deg<<<cdiv(E_, 256), 256, 0, stream>>>(ei + E_, ea, cnt, easum, E_);
  k_loopdiv<<<cdiv(Nn, 256), 256, 0, stream>>>(easum, cnt, lpe, Nn);

  // ---- layer 1: K=3 linear (VALU), attention, aggregate ----
  k_lin3<<<cdiv((long)Nn * D1, 256), 256, 0, stream>>>(x, Wl1, bl1, xl1, Nn, D1);
  k_lin3<<<cdiv((long)Nn * D1, 256), 256, 0, stream>>>(x, Wr1, br1, xr1, Nn, D1);
  zero(lmk, (long)Nn * kH); zero(den, (long)Nn * kH); zero(acc1, (long)Nn * D1);
  k_edge_logits<<<ewb, 256, 0, stream>>>(xl1, xr1, ei, ea, lpe, We1, at1,
                                         logit, lmk, E_, Et, C1);
  k_edge_expsum<<<cdiv(nwav, 256), 256, 0, stream>>>(ei, logit, lmk, den, E_, Et);
  k_edge_scatter<<<ewb, 256, 0, stream>>>(xl1, ei, ea, lpe, logit, den, acc1,
                                          E_, Et, C1);
  k_bias_relu<<<cdiv((long)Nn * D1, 256), 256, 0, stream>>>(acc1, b1, (long)Nn * D1, D1);

  // ---- layer 2: WMMA GEMMs for lin_l / lin_r ----
  k_f2h<<<cdiv((long)Nn * D1, 256), 256, 0, stream>>>(acc1, h1h, (long)Nn * D1);
  k_f2h_t<<<cdiv((long)D1 * D2, 256), 256, 0, stream>>>(Wl2, Wl2h, D1, D2);
  k_f2h_t<<<cdiv((long)D1 * D2, 256), 256, 0, stream>>>(Wr2, Wr2h, D1, D2);
  k_f2h_t<<<cdiv((long)1024 * 1024, 256), 256, 0, stream>>>(Wp1, Wp1h, 1024, 1024);
  k_f2h_t<<<cdiv((long)1024 * 1024, 256), 256, 0, stream>>>(Wp2, Wp2h, 1024, 1024);

  dim3 gg(D2 / 128, cdiv(Nn, 64));
  k_wmma_gemm<<<gg, 256, 0, stream>>>(h1h, Wl2h, bl2, xl2, Nn, D2, D1, 0);
  k_wmma_gemm<<<gg, 256, 0, stream>>>(h1h, Wr2h, br2, xr2, Nn, D2, D1, 0);

  zero(lmk, (long)Nn * kH); zero(den, (long)Nn * kH); zero(acc2, (long)Nn * D2);
  k_edge_logits<<<ewb, 256, 0, stream>>>(xl2, xr2, ei, ea, lpe, We2, at2,
                                         logit, lmk, E_, Et, C2);
  k_edge_expsum<<<cdiv(nwav, 256), 256, 0, stream>>>(ei, logit, lmk, den, E_, Et);
  k_edge_scatter<<<ewb, 256, 0, stream>>>(xl2, ei, ea, lpe, logit, den, acc2,
                                          E_, Et, C2);
  k_bias_relu<<<cdiv((long)Nn * D2, 256), 256, 0, stream>>>(acc2, b2, (long)Nn * D2, D2);

  // ---- global mean pool (8 graphs) ----
  zero(gsum, (long)NG * D2); zero(gcnt, NG);
  k_pool_cnt<<<cdiv(Nn, 256), 256, 0, stream>>>(bat, gcnt, Nn);
  k_pool_sum<<<cdiv((long)Nn * D2, 256), 256, 0, stream>>>(acc2, bat, gsum, Nn, D2);
  k_pool_divpad<<<cdiv(16 * D2, 256), 256, 0, stream>>>(gsum, gcnt, g16, NG, D2);

  // ---- projection MLP on WMMA (M padded 8 -> 16) + L2 normalize ----
  dim3 gp(EMB / 128, 1);
  k_wmma_gemm<<<gp, 256, 0, stream>>>(g16, Wp1h, bp1, t1, 16, EMB, 1024, 1);
  k_f2h<<<cdiv(16L * EMB, 256), 256, 0, stream>>>(t1, t1h, 16L * EMB);
  k_wmma_gemm<<<gp, 256, 0, stream>>>(t1h, Wp2h, bp2, embb, 16, EMB, 1024, 0);
  k_normalize<<<1, 256, 0, stream>>>(embb, (float*)d_out, NG, EMB);
}
